// SKConv2d_36953898615347
// MI455X (gfx1250) — matrix operand order repaired and checked by reading, not compiled, not run on hardware
//
#include <hip/hip_runtime.h>
#include <hip/hip_bf16.h>

typedef __bf16 bf16_t;
typedef __attribute__((ext_vector_type(16))) __bf16 v16bf;
typedef __attribute__((ext_vector_type(8)))  __bf16 v8bf;
typedef __attribute__((ext_vector_type(8)))  float  v8f;
typedef __attribute__((ext_vector_type(4)))  float  v4f;

#define CIN   256
#define COUT  256
#define HS    56
#define WS    56
#define NTAP  9
#define LPITCH 264   // 256 + 8 pad: lane stride 528B = 132 dwords == 4 banks mod 64 -> conflict-free
#define WSTEP 24     // tile start stride: tiles at w0=0 and w0=24, both fully inside W=56
#define NWPOS 34     // 32 pixels + 2 halo

// ---------------------------------------------------------------------------
// Kernel 1: fold both sketch branches into one combined 3x3 kernel, bf16.
// Wcb[tap][o][ci] = ( sum_{s<48} S1s[s,ci,tap]*U1s[s,o]
//                  + sum_{u<432} U2s[u,ci,tap]*S2s[u,o] ) / 12
// Layout [tap][o][ci] => WMMA B fragment is two 16B chunks per lane (K 8h.. and 16+8h..).
// ---------------------------------------------------------------------------
__global__ __launch_bounds__(256) void combine_weights_kernel(
    const float* __restrict__ S1s, const float* __restrict__ S2s,
    const float* __restrict__ U1s, const float* __restrict__ U2s,
    bf16_t* __restrict__ Wcb)
{
  int gid = blockIdx.x * 256 + threadIdx.x;   // 9*256*256 threads, ci fastest
  int ci  = gid & 255;
  int o   = (gid >> 8) & 255;
  int tap = gid >> 16;                        // 0..8

  float acc = 0.f;
#pragma unroll 4
  for (int s = 0; s < 48; ++s)
    acc += S1s[(s * CIN + ci) * NTAP + tap] * U1s[s * COUT + o];
#pragma unroll 4
  for (int u = 0; u < 432; ++u)
    acc += U2s[(u * CIN + ci) * NTAP + tap] * S2s[u * COUT + o];

  Wcb[(tap * COUT + o) * CIN + ci] = (bf16_t)(acc * (1.0f / 12.0f));
}

// ---------------------------------------------------------------------------
// Kernel 2: 3x3 conv (Cin=256 -> Cout=256) as implicit GEMM via WMMA bf16.
// Block = 256 threads (8 waves). Block tile: 32 pixels x 256 couts.
// Each wave: 2 M-tiles (16 pixels) x 2 N-tiles (16 couts) = 4 C tiles.
// A fragments reused across N-tiles, B fragments across M-tiles.
// Tiles start at w0 = 0 / 24: overlap columns are recomputed identically,
// so no wasted lanes and a fully unguarded, vectorized epilogue.
// ---------------------------------------------------------------------------
__global__ __launch_bounds__(256) void skconv_wmma_kernel(
    const float* __restrict__ x, const bf16_t* __restrict__ Wcb,
    const float* __restrict__ bias, float* __restrict__ out)
{
  __shared__ bf16_t xs[3 * NWPOS * LPITCH];   // ~54 KB: 3 rows x 34 wpos x 256 cin (padded)

  const int tid = threadIdx.x;
  const int w0  = blockIdx.x * WSTEP;         // 0 or 24; w0+31 <= 55 always
  const int h   = blockIdx.y;
  const int b   = blockIdx.z;

  // ---- stage x[b, :, h-1..h+1, w0-1..w0+32] into LDS as bf16, zero-padded ----
  for (int e = tid; e < 3 * NWPOS * CIN; e += 256) {
    int row = e / (NWPOS * CIN);
    int rem = e - row * (NWPOS * CIN);
    int ci  = rem / NWPOS;
    int jj  = rem - ci * NWPOS;               // lanes sweep jj -> contiguous global reads
    int hh  = h + row - 1;
    int ww  = w0 - 1 + jj;
    float v = 0.f;
    if (hh >= 0 && hh < HS && ww >= 0 && ww < WS)
      v = x[((b * CIN + ci) * HS + hh) * WS + ww];
    xs[(row * NWPOS + jj) * LPITCH + ci] = (bf16_t)v;
  }
  __syncthreads();

  const int lane  = tid & 31;
  const int wave  = tid >> 5;
  const int half  = lane >> 4;                // 0: lanes 0-15, 1: lanes 16-31
  const int m     = lane & 15;                // pixel row of A / column of B
  const int obase = wave * 32;                // this wave's 32 output channels

  v8f acc00 = {};  // (Mtile0, Ntile0)
  v8f acc01 = {};  // (Mtile0, Ntile1)
  v8f acc10 = {};  // (Mtile1, Ntile0)
  v8f acc11 = {};  // (Mtile1, Ntile1)

#pragma unroll
  for (int tap = 0; tap < NTAP; ++tap) {
    const int dh = tap / 3;
    const int dw = tap - dh * 3;
    // A rows: pixel (h+dh-1, w0 + t*16 + m + dw - 1) == LDS[row=dh][wpos=t*16+m+dw]
    const bf16_t* arow0 = xs + (dh * NWPOS + (m + dw)) * LPITCH;
    const bf16_t* arow1 = arow0 + 16 * LPITCH;
    // B columns: Wcb[tap][obase + m (+16)][...], per-lane K window by `half`
    const bf16_t* wb0 = Wcb + ((size_t)(tap * COUT + obase + m)) * CIN + half * 8;
    const bf16_t* wb1 = wb0 + 16 * CIN;

    // pull next tap's B rows toward L2/L0 early (global_prefetch_b8)
    if (tap < NTAP - 1) {
      __builtin_prefetch(wb0 + (size_t)COUT * CIN, 0, 1);
      __builtin_prefetch(wb1 + (size_t)COUT * CIN, 0, 1);
    }

#pragma unroll
    for (int kc = 0; kc < 8; ++kc) {          // 8 chunks of K=32 over Cin=256
      const int k0 = kc * 32;

      // A fragments (ISA 16-bit A 16x32 layout):
      //   lanes 0-15:  elems 0-7 = K 0..7,  elems 8-15 = K 16..23
      //   lanes 16-31: elems 0-7 = K 8..15, elems 8-15 = K 24..31
      v8bf a0lo = *(const v8bf*)(arow0 + k0 + half * 8);
      v8bf a0hi = *(const v8bf*)(arow0 + k0 + half * 8 + 16);
      v16bf afrag0 = __builtin_shufflevector(a0lo, a0hi,
          0, 1, 2, 3, 4, 5, 6, 7, 8, 9, 10, 11, 12, 13, 14, 15);
      v8bf a1lo = *(const v8bf*)(arow1 + k0 + half * 8);
      v8bf a1hi = *(const v8bf*)(arow1 + k0 + half * 8 + 16);
      v16bf afrag1 = __builtin_shufflevector(a1lo, a1hi,
          0, 1, 2, 3, 4, 5, 6, 7, 8, 9, 10, 11, 12, 13, 14, 15);

      // B fragments (ISA 16-bit B 32x16 layout): same K order as A (8h+s, 16+8h+s)
      v8bf b0lo = *(const v8bf*)(wb0 + k0);
      v8bf b0hi = *(const v8bf*)(wb0 + k0 + 16);
      v16bf bfrag0 = __builtin_shufflevector(b0lo, b0hi,
          0, 1, 2, 3, 4, 5, 6, 7, 8, 9, 10, 11, 12, 13, 14, 15);
      v8bf b1lo = *(const v8bf*)(wb1 + k0);
      v8bf b1hi = *(const v8bf*)(wb1 + k0 + 16);
      v16bf bfrag1 = __builtin_shufflevector(b1lo, b1hi,
          0, 1, 2, 3, 4, 5, 6, 7, 8, 9, 10, 11, 12, 13, 14, 15);

      // 4 independent accumulators -> back-to-back WMMAs with no RAW hazards
      acc00 = __builtin_amdgcn_wmma_f32_16x16x32_bf16(
          false, afrag0, false, bfrag0, (short)0, acc00, false, false);
      acc01 = __builtin_amdgcn_wmma_f32_16x16x32_bf16(
          false, afrag0, false, bfrag1, (short)0, acc01, false, false);
      acc10 = __builtin_amdgcn_wmma_f32_16x16x32_bf16(
          false, afrag1, false, bfrag0, (short)0, acc10, false, false);
      acc11 = __builtin_amdgcn_wmma_f32_16x16x32_bf16(
          false, afrag1, false, bfrag1, (short)0, acc11, false, false);
    }
  }

  // ---- epilogue: C layout lane L, VGPR j -> M = j + 8*half, N = L&15 ----
  // Each lane owns 8 consecutive, always-in-range w positions per tile:
  // store as two float4 -> global_store_b128 (16B alignment guaranteed).
  const int o0 = obase + m;
  const int o1 = obase + 16 + m;
  const float bv0 = bias[o0];
  const float bv1 = bias[o1];
  float* out0 = out + (((size_t)b * COUT + o0) * HS + h) * WS;
  float* out1 = out + (((size_t)b * COUT + o1) * HS + h) * WS;

#pragma unroll
  for (int t = 0; t < 2; ++t) {
    const v8f& a0 = t ? acc10 : acc00;
    const v8f& a1 = t ? acc11 : acc01;
    const int wbase = w0 + t * 16 + 8 * half;  // max 24+16+8 = 48; 48+7 = 55 < 56
    v4f s0lo = { a0[0] + bv0, a0[1] + bv0, a0[2] + bv0, a0[3] + bv0 };
    v4f s0hi = { a0[4] + bv0, a0[5] + bv0, a0[6] + bv0, a0[7] + bv0 };
    v4f s1lo = { a1[0] + bv1, a1[1] + bv1, a1[2] + bv1, a1[3] + bv1 };
    v4f s1hi = { a1[4] + bv1, a1[5] + bv1, a1[6] + bv1, a1[7] + bv1 };
    *(v4f*)(out0 + wbase)     = s0lo;
    *(v4f*)(out0 + wbase + 4) = s0hi;
    *(v4f*)(out1 + wbase)     = s1lo;
    *(v4f*)(out1 + wbase + 4) = s1hi;
  }
}

// ---------------------------------------------------------------------------
extern "C" void kernel_launch(void* const* d_in, const int* in_sizes, int n_in,
                              void* d_out, int out_size, void* d_ws, size_t ws_size,
                              hipStream_t stream) {
  (void)in_sizes; (void)n_in; (void)out_size; (void)ws_size;
  const float* x    = (const float*)d_in[0];
  const float* S1s  = (const float*)d_in[1];
  const float* S2s  = (const float*)d_in[2];
  const float* U1s  = (const float*)d_in[3];
  const float* U2s  = (const float*)d_in[4];
  const float* bias = (const float*)d_in[5];

  bf16_t* Wcb = (bf16_t*)d_ws;                // 9*256*256*2 = 1.18 MB scratch

  // 9*256*256 = 589,824 threads -> 2304 blocks
  combine_weights_kernel<<<2304, 256, 0, stream>>>(S1s, S2s, U1s, U2s, Wcb);

  dim3 grid(2, HS, 32);                       // w-tiles (w0 = 0, 24) x H x B
  skconv_wmma_kernel<<<grid, 256, 0, stream>>>(x, Wcb, bias, (float*)d_out);
}
